// GAT_66821101191795
// MI455X (gfx1250) — compile-verified, matrix operand order checked by reading, not run on hardware
//
#include <hip/hip_runtime.h>

#define N_CONS  10000
#define N_COLS  10000
#define N_NODES 20000
#define N_EDGES 200000
#define NEG_SLOPE 0.2f

typedef __attribute__((ext_vector_type(16))) __bf16 bf16x16;
typedef __attribute__((ext_vector_type(8)))  float  f32x8;

union FragBF16 {
    bf16x16 v;
    uint4   u[2];
};

__device__ __forceinline__ unsigned short f32_to_bf16(float f) {
    unsigned int u = __float_as_uint(f);
    u += 0x7FFFu + ((u >> 16) & 1u);   // round to nearest even
    return (unsigned short)(u >> 16);
}

__device__ __forceinline__ float lrelu(float x) { return x > 0.f ? x : NEG_SLOPE * x; }

// Ordered-float max via integer atomics (works for all finite floats).
__device__ __forceinline__ void atomicMaxF32(float* addr, float v) {
    if (v >= 0.f) atomicMax((int*)addr, __float_as_int(v));
    else          atomicMin((unsigned int*)addr, __float_as_uint(v));
}

// ---------------------------------------------------------------------------
// f32 -> bf16 elementwise convert
// ---------------------------------------------------------------------------
__global__ void cvt_f32_bf16(const float* __restrict__ in,
                             unsigned short* __restrict__ out, int n) {
    int i = blockIdx.x * blockDim.x + threadIdx.x;
    if (i < n) out[i] = f32_to_bf16(in[i]);
}

// ---------------------------------------------------------------------------
// Initial embedding: feature tile x2, tiny dense layer, ReLU, store bf16.
// ---------------------------------------------------------------------------
__global__ void embed_kernel(const float* __restrict__ cons, const float* __restrict__ cols,
                             const float* __restrict__ nodeW, const float* __restrict__ nodeB,
                             const float* __restrict__ colW, const float* __restrict__ colB,
                             unsigned short* __restrict__ out) {
    int idx = blockIdx.x * blockDim.x + threadIdx.x;
    if (idx >= N_NODES * 128) return;
    int n = idx >> 7, c = idx & 127;
    float acc;
    if (n < N_CONS) {
        acc = nodeB[c];
        const float* w = nodeW + c * 8;
        const float* x = cons + (size_t)n * 4;
        #pragma unroll
        for (int j = 0; j < 8; ++j) acc += x[j & 3] * w[j];      // tile(x, (1,2))
    } else {
        int m = n - N_CONS;
        acc = colB[c];
        const float* w = colW + c * 16;
        const float* x = cols + (size_t)m * 8;
        #pragma unroll
        for (int j = 0; j < 16; ++j) acc += x[j & 7] * w[j];
    }
    out[idx] = f32_to_bf16(fmaxf(acc, 0.f));
}

// ---------------------------------------------------------------------------
// WMMA GEMM:  C[M,N] (f32) = A[M,K] (bf16, row-major) * B[N,K]^T (bf16, row-major)
// One wave computes a 16x64 strip; A fragment reused across 4 WMMAs.
// Requires M%16==0, N%64==0, K%32==0 (all shapes here comply).
// ---------------------------------------------------------------------------
__global__ void gemm_bf16_nt(const unsigned short* __restrict__ A,
                             const unsigned short* __restrict__ B,
                             const float* __restrict__ bias,
                             float* __restrict__ C,
                             int M, int N, int K) {
    const int wave = (blockIdx.x * blockDim.x + threadIdx.x) >> 5;
    const int lane = threadIdx.x & 31;
    const int ntiles = N >> 6;
    const int mt = wave / ntiles;
    const int nt = wave - mt * ntiles;
    if (mt >= (M >> 4)) return;                    // wave-uniform guard
    const int row0 = mt << 4;
    const int col0 = nt << 6;
    const int r    = lane & 15;
    const int koff = (lane >> 4) << 3;             // 0 or 8 (ISA bf16 A/B layout)

    f32x8 acc0 = {}, acc1 = {}, acc2 = {}, acc3 = {};
    const unsigned short* ap = A + (size_t)(row0 + r) * K + koff;
    const unsigned short* bp = B + (size_t)(col0 + r) * K + koff;
    const size_t bstride = (size_t)16 * K;

    for (int k = 0; k < K; k += 32) {
        FragBF16 fa, fb0, fb1, fb2, fb3;
        fa.u[0]  = *(const uint4*)(ap + k);
        fa.u[1]  = *(const uint4*)(ap + k + 16);
        fb0.u[0] = *(const uint4*)(bp + k);
        fb0.u[1] = *(const uint4*)(bp + k + 16);
        fb1.u[0] = *(const uint4*)(bp + bstride + k);
        fb1.u[1] = *(const uint4*)(bp + bstride + k + 16);
        fb2.u[0] = *(const uint4*)(bp + 2 * bstride + k);
        fb2.u[1] = *(const uint4*)(bp + 2 * bstride + k + 16);
        fb3.u[0] = *(const uint4*)(bp + 3 * bstride + k);
        fb3.u[1] = *(const uint4*)(bp + 3 * bstride + k + 16);
        acc0 = __builtin_amdgcn_wmma_f32_16x16x32_bf16(false, fa.v, false, fb0.v, (short)0, acc0, false, false);
        acc1 = __builtin_amdgcn_wmma_f32_16x16x32_bf16(false, fa.v, false, fb1.v, (short)0, acc1, false, false);
        acc2 = __builtin_amdgcn_wmma_f32_16x16x32_bf16(false, fa.v, false, fb2.v, (short)0, acc2, false, false);
        acc3 = __builtin_amdgcn_wmma_f32_16x16x32_bf16(false, fa.v, false, fb3.v, (short)0, acc3, false, false);
    }

    // C/D layout: VGPR i, lanes 0-15 -> M=i, lanes 16-31 -> M=8+i; N = lane&15.
    const int cn = lane & 15;
    const int cm = (lane >> 4) << 3;
    float bb0 = 0.f, bb1 = 0.f, bb2 = 0.f, bb3 = 0.f;
    if (bias) {
        bb0 = bias[col0 + cn];      bb1 = bias[col0 + 16 + cn];
        bb2 = bias[col0 + 32 + cn]; bb3 = bias[col0 + 48 + cn];
    }
    float* crow = C + (size_t)(row0 + cm) * N + col0 + cn;
    #pragma unroll
    for (int i = 0; i < 8; ++i) {
        crow[0]  = acc0[i] + bb0;
        crow[16] = acc1[i] + bb1;
        crow[32] = acc2[i] + bb2;
        crow[48] = acc3[i] + bb3;
        crow += N;
    }
}

// ---------------------------------------------------------------------------
// Attention scores: one wave per (node, head); coalesced 128-ch dot products.
// ---------------------------------------------------------------------------
__global__ void att_scores(const float* __restrict__ xp, const float* __restrict__ attS,
                           const float* __restrict__ attD, float* __restrict__ aS,
                           float* __restrict__ aD, int nNodes, int heads) {
    int wid  = (blockIdx.x * blockDim.x + threadIdx.x) >> 5;
    int lane = threadIdx.x & 31;
    if (wid >= nNodes * heads) return;
    int h = wid % heads;
    const float* xrow = xp + (size_t)wid * 128;
    const float* as   = attS + h * 128;
    const float* ad   = attD + h * 128;
    float s = 0.f, d = 0.f;
    for (int c = lane; c < 128; c += 32) {
        float v = xrow[c];
        s += v * as[c];
        d += v * ad[c];
    }
    #pragma unroll
    for (int off = 16; off; off >>= 1) {
        s += __shfl_xor(s, off, 32);
        d += __shfl_xor(d, off, 32);
    }
    if (lane == 0) { aS[wid] = s; aD[wid] = d; }
}

// ---------------------------------------------------------------------------
// Segment softmax pieces (self loop included via init).
// ---------------------------------------------------------------------------
__global__ void init_m(const float* __restrict__ aS, const float* __restrict__ aD,
                       float* __restrict__ m, int count) {
    int i = blockIdx.x * blockDim.x + threadIdx.x;
    if (i < count) m[i] = lrelu(aS[i] + aD[i]);                  // self-loop score
}

__global__ void edge_max(const int* __restrict__ edges, const float* __restrict__ aS,
                         const float* __restrict__ aD, float* __restrict__ m,
                         int E, int heads) {
    int idx = blockIdx.x * blockDim.x + threadIdx.x;
    if (idx >= E * heads) return;
    int e = idx / heads, h = idx - e * heads;
    int sN = edges[e], dN = edges[E + e];
    float v = lrelu(aS[sN * heads + h] + aD[dN * heads + h]);
    atomicMaxF32(m + dN * heads + h, v);
}

__global__ void init_s(const float* __restrict__ aS, const float* __restrict__ aD,
                       const float* __restrict__ m, float* __restrict__ s, int count) {
    int i = blockIdx.x * blockDim.x + threadIdx.x;
    if (i < count) s[i] = expf(lrelu(aS[i] + aD[i]) - m[i]);     // self-loop term
}

__global__ void edge_sum(const int* __restrict__ edges, const float* __restrict__ aS,
                         const float* __restrict__ aD, const float* __restrict__ m,
                         float* __restrict__ s, int E, int heads) {
    int idx = blockIdx.x * blockDim.x + threadIdx.x;
    if (idx >= E * heads) return;
    int e = idx / heads, h = idx - e * heads;
    int sN = edges[e], dN = edges[E + e];
    int dnh = dN * heads + h;
    float v = lrelu(aS[sN * heads + h] + aD[dnh]);
    atomicAdd(s + dnh, expf(v - m[dnh]));
}

// Initialize aggregation with the self-loop contribution (also inits buffer).
__global__ void agg_self(const float* __restrict__ xp, const float* __restrict__ aS,
                         const float* __restrict__ aD, const float* __restrict__ m,
                         const float* __restrict__ s, float* __restrict__ agg,
                         int nNodes, int heads) {
    int idx = blockIdx.x * blockDim.x + threadIdx.x;
    if (idx >= nNodes * heads * 32) return;
    int c4 = idx & 31;
    int nh = idx >> 5;
    float w = expf(lrelu(aS[nh] + aD[nh]) - m[nh]) / (s[nh] + 1e-16f);
    float4 v = ((const float4*)xp)[(size_t)nh * 32 + c4];
    float4 o; o.x = v.x * w; o.y = v.y * w; o.z = v.z * w; o.w = v.w * w;
    ((float4*)agg)[(size_t)nh * 32 + c4] = o;
}

// Per-edge weighted scatter-add (L2-resident atomics).
__global__ void edge_agg(const int* __restrict__ edges, const float* __restrict__ xp,
                         const float* __restrict__ aS, const float* __restrict__ aD,
                         const float* __restrict__ m, const float* __restrict__ s,
                         float* __restrict__ agg, int E, int heads) {
    int idx = blockIdx.x * blockDim.x + threadIdx.x;
    if (idx >= E * heads * 32) return;
    int c4 = idx & 31;
    int eh = idx >> 5;
    int h  = eh % heads;
    int e  = eh / heads;
    int sN = edges[e], dN = edges[E + e];
    int snh = sN * heads + h;
    int dnh = dN * heads + h;
    float ev = lrelu(aS[snh] + aD[dnh]);
    float w  = expf(ev - m[dnh]) / (s[dnh] + 1e-16f);
    float4 v = ((const float4*)xp)[(size_t)snh * 32 + c4];
    float* o = agg + (size_t)dnh * 128 + (c4 << 2);
    atomicAdd(o + 0, v.x * w);
    atomicAdd(o + 1, v.y * w);
    atomicAdd(o + 2, v.z * w);
    atomicAdd(o + 3, v.w * w);
}

// Fused bias + ReLU + bf16 convert.
__global__ void relu_bias_bf16(const float* __restrict__ in, const float* __restrict__ bias,
                               unsigned short* __restrict__ out, int total, int ld) {
    int idx = blockIdx.x * blockDim.x + threadIdx.x;
    if (idx >= total) return;
    int c = idx % ld;
    out[idx] = f32_to_bf16(fmaxf(in[idx] + bias[c], 0.f));
}

// ---------------------------------------------------------------------------
extern "C" void kernel_launch(void* const* d_in, const int* in_sizes, int n_in,
                              void* d_out, int out_size, void* d_ws, size_t ws_size,
                              hipStream_t stream) {
    (void)in_sizes; (void)n_in; (void)out_size; (void)ws_size;

    const float* cons  = (const float*)d_in[0];
    const float* colsX = (const float*)d_in[1];
    const float* nodeW = (const float*)d_in[2];
    const float* nodeB = (const float*)d_in[3];
    const float* colW  = (const float*)d_in[4];
    const float* colB  = (const float*)d_in[5];
    const float* W1    = (const float*)d_in[6];
    const float* attS1 = (const float*)d_in[7];
    const float* attD1 = (const float*)d_in[8];
    const float* b1    = (const float*)d_in[9];
    const float* W2    = (const float*)d_in[10];
    const float* attS2 = (const float*)d_in[11];
    const float* attD2 = (const float*)d_in[12];
    const float* b2    = (const float*)d_in[13];
    const float* outW  = (const float*)d_in[14];
    const float* outB  = (const float*)d_in[15];
    const int*   edges = (const int*)d_in[16];

    char* ws = (char*)d_ws;
    size_t off = 0;
    auto take = [&](size_t bytes) {
        char* p = ws + off;
        off += (bytes + 255) & ~(size_t)255;
        return p;
    };

    unsigned short* W1b   = (unsigned short*)take((size_t)1024 * 128 * 2);
    unsigned short* W2b   = (unsigned short*)take((size_t)128 * 1024 * 2);
    unsigned short* oWb   = (unsigned short*)take((size_t)128 * 128 * 2);
    unsigned short* emb0b = (unsigned short*)take((size_t)N_NODES * 128 * 2);
    float* xp1            = (float*)take((size_t)N_NODES * 1024 * 4);
    float* aS1            = (float*)take((size_t)N_NODES * 8 * 4);
    float* aD1            = (float*)take((size_t)N_NODES * 8 * 4);
    float* m1             = (float*)take((size_t)N_NODES * 8 * 4);
    float* s1             = (float*)take((size_t)N_NODES * 8 * 4);
    float* agg1           = (float*)take((size_t)N_NODES * 1024 * 4);
    unsigned short* emb1b = (unsigned short*)take((size_t)N_NODES * 1024 * 2);
    float* xp2            = (float*)take((size_t)N_NODES * 128 * 4);
    float* aS2            = (float*)take((size_t)N_NODES * 4);
    float* aD2            = (float*)take((size_t)N_NODES * 4);
    float* m2             = (float*)take((size_t)N_NODES * 4);
    float* s2             = (float*)take((size_t)N_NODES * 4);
    float* agg2           = (float*)take((size_t)N_NODES * 128 * 4);
    unsigned short* emb2b = (unsigned short*)take((size_t)N_NODES * 128 * 2);

    const dim3 B256(256);
    auto blocks = [](long long t) { return dim3((unsigned)((t + 255) / 256)); };

    // Weight conversion to bf16.
    cvt_f32_bf16<<<blocks(1024 * 128), B256, 0, stream>>>(W1, W1b, 1024 * 128);
    cvt_f32_bf16<<<blocks(128 * 1024), B256, 0, stream>>>(W2, W2b, 128 * 1024);
    cvt_f32_bf16<<<blocks(128 * 128), B256, 0, stream>>>(outW, oWb, 128 * 128);

    // Initial embedding (fused tile + linear + ReLU) -> bf16.
    embed_kernel<<<blocks((long long)N_NODES * 128), B256, 0, stream>>>(
        cons, colsX, nodeW, nodeB, colW, colB, emb0b);

    // ---- GAT layer 1 (heads=8, out_ch=128) ----
    {   int nw = (N_NODES / 16) * (1024 / 64);    // 20000 waves
        gemm_bf16_nt<<<blocks((long long)nw * 32), B256, 0, stream>>>(
            emb0b, W1b, nullptr, xp1, N_NODES, 1024, 128);
    }
    att_scores<<<blocks((long long)N_NODES * 8 * 32), B256, 0, stream>>>(
        xp1, attS1, attD1, aS1, aD1, N_NODES, 8);
    init_m<<<blocks(N_NODES * 8), B256, 0, stream>>>(aS1, aD1, m1, N_NODES * 8);
    edge_max<<<blocks((long long)N_EDGES * 8), B256, 0, stream>>>(edges, aS1, aD1, m1, N_EDGES, 8);
    init_s<<<blocks(N_NODES * 8), B256, 0, stream>>>(aS1, aD1, m1, s1, N_NODES * 8);
    edge_sum<<<blocks((long long)N_EDGES * 8), B256, 0, stream>>>(edges, aS1, aD1, m1, s1, N_EDGES, 8);
    agg_self<<<blocks((long long)N_NODES * 8 * 32), B256, 0, stream>>>(
        xp1, aS1, aD1, m1, s1, agg1, N_NODES, 8);
    edge_agg<<<blocks((long long)N_EDGES * 8 * 32), B256, 0, stream>>>(
        edges, xp1, aS1, aD1, m1, s1, agg1, N_EDGES, 8);
    relu_bias_bf16<<<blocks((long long)N_NODES * 1024), B256, 0, stream>>>(
        agg1, b1, emb1b, N_NODES * 1024, 1024);

    // ---- GAT layer 2 (heads=1, out_ch=128; same edge set) ----
    {   int nw = (N_NODES / 16) * (128 / 64);     // 2500 waves
        gemm_bf16_nt<<<blocks((long long)nw * 32), B256, 0, stream>>>(
            emb1b, W2b, nullptr, xp2, N_NODES, 128, 1024);
    }
    att_scores<<<blocks((long long)N_NODES * 32), B256, 0, stream>>>(
        xp2, attS2, attD2, aS2, aD2, N_NODES, 1);
    init_m<<<blocks(N_NODES), B256, 0, stream>>>(aS2, aD2, m2, N_NODES);
    edge_max<<<blocks(N_EDGES), B256, 0, stream>>>(edges, aS2, aD2, m2, N_EDGES, 1);
    init_s<<<blocks(N_NODES), B256, 0, stream>>>(aS2, aD2, m2, s2, N_NODES);
    edge_sum<<<blocks(N_EDGES), B256, 0, stream>>>(edges, aS2, aD2, m2, s2, N_EDGES, 1);
    agg_self<<<blocks((long long)N_NODES * 32), B256, 0, stream>>>(
        xp2, aS2, aD2, m2, s2, agg2, N_NODES, 1);
    edge_agg<<<blocks((long long)N_EDGES * 32), B256, 0, stream>>>(
        edges, xp2, aS2, aD2, m2, s2, agg2, N_EDGES, 1);
    relu_bias_bf16<<<blocks((long long)N_NODES * 128), B256, 0, stream>>>(
        agg2, b2, emb2b, N_NODES * 128, 128);

    // ---- Output head: logits = emb2[N_CONS:] @ out_W^T + out_b -> d_out ----
    {   int nw = (N_COLS / 16) * (128 / 64);      // 1250 waves
        gemm_bf16_nt<<<blocks((long long)nw * 32), B256, 0, stream>>>(
            emb2b + (size_t)N_CONS * 128, oWb, outB, (float*)d_out, N_COLS, 128, 128);
    }
}
